// YvSSDuality_72112500900649
// MI455X (gfx1250) — compile-verified, matrix operand order checked by reading, not run on hardware
//
#include <hip/hip_runtime.h>
#include <hip/hip_bf16.h>

// ---------------------------------------------------------------------------
// Selective-scan (Mamba SSM) for MI455X / gfx1250, wave32.
//   B=4, S=256, D=4096, N=16, R=256
// Pipeline:
//   k0: pack/convert f32->f16 operands (delta, Wdt, x, WB|WC)
//   k1: WMMA GEMM  dt_raw = delta @ Wdt^T  (M=1024,N=4096,K=256), fused
//       softplus+clamp epilogue -> dtbuf (f32)
//   k2: WMMA GEMM  [Bm|Cm] = x @ [WB;WC]^T (M=1024,N=32,K=4096)
//   k3: scan kernel: 1 thread per (b,d) channel, h[16] in VGPRs,
//       Bm/Cm slab DMA'd into LDS via TDM (tensor_load_to_lds +
//       s_wait_tensorcnt), exp via v_exp_f32, prefetch next row.
// ---------------------------------------------------------------------------

#define B_ 4
#define S_ 256
#define D_ 4096
#define N_ 16
#define R_ 256
#define MROWS (B_ * S_)        // 1024

typedef __attribute__((ext_vector_type(16))) _Float16 v16h;
typedef __attribute__((ext_vector_type(8)))  _Float16 v8h;
typedef __attribute__((ext_vector_type(8)))  float    v8f;
typedef __attribute__((ext_vector_type(4)))  unsigned int v4u;
typedef __attribute__((ext_vector_type(4)))  int      v4i;
typedef __attribute__((ext_vector_type(8)))  int      v8i;

// ---------------------------------------------------------------------------
// k0: convert/pack all f16 operands in one grid.
// ---------------------------------------------------------------------------
__global__ void k0_pack(const float* __restrict__ delta,
                        const float* __restrict__ Wdt,
                        const float* __restrict__ x,
                        const float* __restrict__ WB,
                        const float* __restrict__ WC,
                        _Float16* __restrict__ deltah,
                        _Float16* __restrict__ Wdth,
                        _Float16* __restrict__ xh,
                        _Float16* __restrict__ Wbch) {
  const int n0 = MROWS * R_;          // 262144
  const int n1 = D_ * R_;             // 1048576
  const int n2 = MROWS * D_;          // 4194304
  const int n3 = 32 * D_;             // 131072
  int i = blockIdx.x * blockDim.x + threadIdx.x;
  if (i < n0) {
    deltah[i] = (_Float16)delta[i];
  } else if (i < n0 + n1) {
    int j = i - n0;
    Wdth[j] = (_Float16)Wdt[j];
  } else if (i < n0 + n1 + n2) {
    int j = i - n0 - n1;
    xh[j] = (_Float16)x[j];
  } else if (i < n0 + n1 + n2 + n3) {
    int j = i - n0 - n1 - n2;
    Wbch[j] = (j < 16 * D_) ? (_Float16)WB[j] : (_Float16)WC[j - 16 * D_];
  }
}

// ---------------------------------------------------------------------------
// Generic per-wave 16x16-tile f16 WMMA GEMM:  out[m][n] = sum_k A[m][k]*Bn[n][k]
//   Ah: [M][K] row-major halves, Bh: [Ncols][K] n-major halves.
// mode 0: softplus+clamp -> out0[m*Ncols + n]  (dt path, +bias)
// mode 1: split          -> n<16: out0[m*16+n] (Bm), n>=16: out1 (Cm)
// ---------------------------------------------------------------------------
__global__ void k1_wmma_gemm(const _Float16* __restrict__ Ah,
                             const _Float16* __restrict__ Bh,
                             const float* __restrict__ bias,   // may be null
                             float* __restrict__ out0,
                             float* __restrict__ out1,
                             int M, int Ncols, int K, int mode) {
  const int wave    = (blockIdx.x * blockDim.x + threadIdx.x) >> 5;
  const int tiles_n = Ncols >> 4;
  const int tiles_m = M >> 4;
  const int tm = wave / tiles_n;
  const int tn = wave % tiles_n;
  if (tm >= tiles_m) return;

  const int lane = threadIdx.x & 31;
  const int half = lane >> 4;       // K-half selector (ISA 16-bit A layout)
  const int lr   = lane & 15;       // row (A) / col (B,C)

  const _Float16* ap = Ah + (size_t)(tm * 16 + lr) * K;
  const _Float16* bp = Bh + (size_t)(tn * 16 + lr) * K;

  v8f acc = {};
  for (int kk = 0; kk < K; kk += 32) {
    const int r0 = kk + 8 * half;        // K-run 0 (8 contiguous halves)
    const int r1 = kk + 16 + 8 * half;   // K-run 1
    v8h a0 = *(const v8h*)(ap + r0);
    v8h a1 = *(const v8h*)(ap + r1);
    v8h b0 = *(const v8h*)(bp + r0);
    v8h b1 = *(const v8h*)(bp + r1);
    v16h a, b;
#pragma unroll
    for (int i = 0; i < 8; ++i) {
      a[i] = a0[i]; a[i + 8] = a1[i];
      b[i] = b0[i]; b[i + 8] = b1[i];
    }
    acc = __builtin_amdgcn_wmma_f32_16x16x32_f16(
        /*neg_a=*/false, a, /*neg_b=*/false, b,
        /*c_mod=*/(short)0, acc, /*reuse_a=*/false, /*reuse_b=*/false);
  }

  const int col = tn * 16 + lr;
  if (mode == 0) {
    const float bv = bias ? bias[col] : 0.0f;
#pragma unroll
    for (int v = 0; v < 8; ++v) {
      const int m = tm * 16 + v + 8 * half;
      float z = acc[v] + bv;
      float sp = (z > 15.0f)
                   ? z
                   : 0.69314718056f *
                     __builtin_amdgcn_logf(1.0f + __builtin_amdgcn_exp2f(z * 1.44269504089f));
      sp = fminf(fmaxf(sp, 1e-6f), 10.0f);
      out0[(size_t)m * Ncols + col] = sp;
    }
  } else {
#pragma unroll
    for (int v = 0; v < 8; ++v) {
      const int m = tm * 16 + v + 8 * half;
      if (col < 16) out0[(size_t)m * 16 + col]        = acc[v];
      else          out1[(size_t)m * 16 + (col - 16)] = acc[v];
    }
  }
}

// ---------------------------------------------------------------------------
// TDM 1-D copy: global -> LDS, nelem f32 elements, one DMA descriptor.
// D# layout per CDNA5 ISA §8.3/§8.4 (groups 2/3 zero: <=2D tensor).
// Issued once per calling wave; EXEC is ignored by TDM.
// This toolchain's builtin is the 6-arg form:
//   (uint32x4 g0, int32x8 g1, int32x4 g2, int32x4 g3, int32x8 extra, i32 cpol)
// ---------------------------------------------------------------------------
__device__ __forceinline__ void tdm_copy_f32(const float* gsrc, float* lds_dst,
                                             int nelem) {
  // addrspacecast(LDS->flat) keeps the 32-bit LDS byte offset in the low word.
  unsigned lds_off = (unsigned)(unsigned long long)(uintptr_t)lds_dst;
  unsigned long long ga = (unsigned long long)(uintptr_t)gsrc;

  v4u g0;
  g0[0] = 1u;                                   // count=1 valid descriptor
  g0[1] = lds_off;                              // lds_addr [63:32]
  g0[2] = (unsigned)(ga & 0xffffffffu);         // global_addr[31:0]
  g0[3] = (unsigned)((ga >> 32) & 0x1ffffffu)   // global_addr[56:32]
          | (2u << 30);                         // type=2 ("image")

  v8i g1;
  g1[0] = (2 << 16);                            // data_size=2 -> 4 bytes
  g1[1] = (nelem & 0xffff) << 16;               // tensor_dim0[15:0] @ [63:48]
  g1[2] = ((nelem >> 16) & 0xffff)              // tensor_dim0[31:16]
          | (1 << 16);                          // tensor_dim1 = 1
  g1[3] = (nelem & 0xffff) << 16;               // tile_dim0 @ [127:112]
  g1[4] = 1;                                    // tile_dim1=1, tile_dim2=0
  g1[5] = nelem;                                // tensor_dim0_stride[31:0]
  g1[6] = 0;                                    // stride hi / dim1_stride lo
  g1[7] = 0;                                    // tensor_dim1_stride hi

  v4i gz4 = {0, 0, 0, 0};                       // groups 2/3: unused dims
  v8i gz8 = {0, 0, 0, 0, 0, 0, 0, 0};           // extra SGPR group (unused)
  __builtin_amdgcn_tensor_load_to_lds(g0, g1, gz4, gz4, gz8, 0);
}

// ---------------------------------------------------------------------------
// k3: recurrent scan. grid = (D_/256, B_), 256 threads (8 wave32).
// Thread owns channel d for batch b; h[16] + A[16] in VGPRs.
// Bm/Cm slab (16 KB each) staged into LDS by the Tensor Data Mover.
// ---------------------------------------------------------------------------
__global__ void k3_scan(const float* __restrict__ x,
                        const float* __restrict__ dtbuf,
                        const float* __restrict__ Bm,
                        const float* __restrict__ Cm,
                        const float* __restrict__ A_log,
                        const float* __restrict__ Dvec,
                        float* __restrict__ out) {
  __shared__ float sB[S_ * N_];
  __shared__ float sC[S_ * N_];

  const int b = blockIdx.y;
  const int d = blockIdx.x * 256 + threadIdx.x;

  // Wave 0 issues two TDM descriptors (EXEC ignored by TDM; one issue/wave),
  // waits TENSORcnt==0, then the block barrier publishes the LDS slab.
  if (threadIdx.x < 32) {
    tdm_copy_f32(Bm + (size_t)b * S_ * N_, sB, S_ * N_);
    tdm_copy_f32(Cm + (size_t)b * S_ * N_, sC, S_ * N_);
    __builtin_amdgcn_s_wait_tensorcnt(0);
  }
  __syncthreads();

  // A = clip(-exp(A_log), -10, -1e-6); pre-scale by log2(e) for v_exp_f32.
  float ALn2[N_];
#pragma unroll
  for (int n = 0; n < N_; ++n) {
    float e = __builtin_amdgcn_exp2f(A_log[(size_t)n * D_ + d] * 1.44269504089f);
    e = fminf(fmaxf(e, 1e-6f), 10.0f);
    ALn2[n] = -e * 1.44269504089f;    // A[n] * log2(e)
  }
  const float Dv = Dvec[d];

  float h[N_];
#pragma unroll
  for (int n = 0; n < N_; ++n) h[n] = 0.0f;

  size_t idx = (size_t)b * S_ * D_ + d;
  for (int s = 0; s < S_; ++s, idx += D_) {
    const float dt = dtbuf[idx];
    const float xv = x[idx];
    // Pull next time-step rows toward the caches (global_prefetch_b8).
    __builtin_prefetch(x + idx + D_, 0, 3);
    __builtin_prefetch(dtbuf + idx + D_, 0, 3);

    const float dtx = dt * xv;
    const float* bs = &sB[s * N_];
    const float* cs = &sC[s * N_];
    float y = 0.0f;
#pragma unroll
    for (int n = 0; n < N_; ++n) {
      const float dA = __builtin_amdgcn_exp2f(dt * ALn2[n]);   // exp(dt*A[n])
      h[n] = fmaf(dA, h[n], dtx * bs[n]);
      y = fmaf(h[n], cs[n], y);
    }
    float o = fmaf(xv, Dv, y);
    o = fminf(fmaxf(o, -1e4f), 1e4f);   // clamp => finite; global NaN-zeroing
                                        // path of the reference is unreachable
    out[idx] = o;
  }
}

// ---------------------------------------------------------------------------
// Host launcher. Workspace layout (all 16B-aligned given 256B-aligned d_ws):
//   dtbuf  f32 [1024*4096]   16 MB
//   Bm     f32 [1024*16]     64 KB
//   Cm     f32 [1024*16]     64 KB
//   deltah f16 [1024*256]    512 KB
//   Wdth   f16 [4096*256]    2 MB
//   xh     f16 [1024*4096]   8 MB
//   Wbch   f16 [32*4096]     256 KB     (~26.9 MB total)
// ---------------------------------------------------------------------------
extern "C" void kernel_launch(void* const* d_in, const int* in_sizes, int n_in,
                              void* d_out, int out_size, void* d_ws, size_t ws_size,
                              hipStream_t stream) {
  const float* x     = (const float*)d_in[0];
  const float* delta = (const float*)d_in[1];
  const float* Wdt   = (const float*)d_in[2];
  const float* bdt   = (const float*)d_in[3];
  const float* A_log = (const float*)d_in[4];
  const float* Dvec  = (const float*)d_in[5];
  const float* WB    = (const float*)d_in[6];
  const float* WC    = (const float*)d_in[7];
  float* out = (float*)d_out;

  float*    dtbuf  = (float*)d_ws;
  float*    Bm     = dtbuf + (size_t)MROWS * D_;
  float*    Cm     = Bm + (size_t)MROWS * N_;
  _Float16* deltah = (_Float16*)(Cm + (size_t)MROWS * N_);
  _Float16* Wdth   = deltah + (size_t)MROWS * R_;
  _Float16* xh     = Wdth + (size_t)D_ * R_;
  _Float16* Wbch   = xh + (size_t)MROWS * D_;

  // k0: pack all f16 operands.
  {
    const int total = MROWS * R_ + D_ * R_ + MROWS * D_ + 32 * D_;
    dim3 grid((total + 255) / 256);
    k0_pack<<<grid, 256, 0, stream>>>(delta, Wdt, x, WB, WC,
                                      deltah, Wdth, xh, Wbch);
  }
  // k1: dt GEMM  (M=1024, N=4096, K=256), softplus epilogue.
  {
    const int waves = (MROWS / 16) * (D_ / 16);      // 16384 tiles
    dim3 grid(waves / 8);                            // 8 waves / block
    k1_wmma_gemm<<<grid, 256, 0, stream>>>(deltah, Wdth, bdt,
                                           dtbuf, nullptr,
                                           MROWS, D_, R_, /*mode=*/0);
  }
  // k2: B/C projections (M=1024, N=32, K=4096).
  {
    const int waves = (MROWS / 16) * (32 / 16);      // 128 tiles
    dim3 grid(waves / 8);                            // 16 blocks
    k1_wmma_gemm<<<grid, 256, 0, stream>>>(xh, Wbch, nullptr,
                                           Bm, Cm,
                                           MROWS, 32, D_, /*mode=*/1);
  }
  // k3: recurrent scan.
  {
    dim3 grid(D_ / 256, B_);
    k3_scan<<<grid, 256, 0, stream>>>(x, dtbuf, Bm, Cm, A_log, Dvec, out);
  }
}
